// CamLiPWC_Core_83786222010988
// MI455X (gfx1250) — matrix operand order repaired and checked by reading, not compiled, not run on hardware
//
#include <hip/hip_runtime.h>
#include <hip/hip_bf16.h>
#include <float.h>

#define B_  2
#define N_  8192
#define M_  2048
#define C_  64
#define H_  160
#define W_  480
#define MD  4
#define ND  9      // 2*MD+1
#define NDISP 81   // ND*ND
#define KS  16

typedef __attribute__((ext_vector_type(16))) _Float16 v16h;
typedef __attribute__((ext_vector_type(8)))  _Float16 v8h;
typedef __attribute__((ext_vector_type(2)))  _Float16 v2h;
typedef __attribute__((ext_vector_type(8)))  float    v8f;

// ---------------------------------------------------------------------------
// Async global->LDS copy (CDNA5): one b32 per lane, tracked by ASYNCcnt.
// ---------------------------------------------------------------------------
__device__ __forceinline__ void async_g2l_b32(unsigned lds_off, const float* gaddr) {
    asm volatile("global_load_async_to_lds_b32 %0, %1, off"
                 :: "v"(lds_off), "v"(gaddr) : "memory");
}
__device__ __forceinline__ void wait_async0() {
    asm volatile("s_wait_asynccnt 0x0" ::: "memory");
}

// ---------------------------------------------------------------------------
// Kernel A: 3-NN inverse-distance interpolation (67ch) + xyz_warp + uv1
// ---------------------------------------------------------------------------
__global__ __launch_bounds__(256) void k_interp(
    const float* __restrict__ xyz1,  const float* __restrict__ xyzc,
    const float* __restrict__ flowc, const float* __restrict__ featc,
    float* __restrict__ out_interp, float* __restrict__ out_warp,
    float* __restrict__ out_uv)
{
    __shared__ float s_pts[3 * M_];
    const int b = blockIdx.x / (N_ / 256);
    const int n = (blockIdx.x % (N_ / 256)) * 256 + threadIdx.x;

    const float* src = xyzc + b * 3 * M_;
    {
        const unsigned base_off = (unsigned)(unsigned long long)(&s_pts[0]);
        for (int i = threadIdx.x; i < 3 * M_; i += 256)
            async_g2l_b32(base_off + 4u * i, src + i);
        wait_async0();
    }
    __syncthreads();

    const float qx = xyz1[(b * 3 + 0) * N_ + n];
    const float qy = xyz1[(b * 3 + 1) * N_ + n];
    const float qz = xyz1[(b * 3 + 2) * N_ + n];

    float d0 = FLT_MAX, d1 = FLT_MAX, d2 = FLT_MAX;
    int   i0 = 0, i1 = 0, i2 = 0;
    const float* px = s_pts;
    const float* py = s_pts + M_;
    const float* pz = s_pts + 2 * M_;
    for (int p = 0; p < M_; ++p) {
        float dx = qx - px[p], dy = qy - py[p], dz = qz - pz[p];
        float dd = dx * dx + dy * dy + dz * dz;
        if (dd < d2) {
            if (dd < d0)      { d2 = d1; i2 = i1; d1 = d0; i1 = i0; d0 = dd; i0 = p; }
            else if (dd < d1) { d2 = d1; i2 = i1; d1 = dd; i1 = p; }
            else              { d2 = dd; i2 = p; }
        }
    }
    float w0 = 1.0f / (sqrtf(d0) + 1e-8f);
    float w1 = 1.0f / (sqrtf(d1) + 1e-8f);
    float w2 = 1.0f / (sqrtf(d2) + 1e-8f);
    const float wsum = w0 + w1 + w2;
    w0 /= wsum; w1 /= wsum; w2 /= wsum;

    for (int c = 0; c < 67; ++c) {
        const float* sf = (c < 3) ? (flowc + (b * 3 + c) * M_)
                                  : (featc + (b * 64 + (c - 3)) * M_);
        float v = w0 * sf[i0] + w1 * sf[i1] + w2 * sf[i2];
        out_interp[(b * 67 + c) * N_ + n] = v;
        if (c < 3) {
            float q = (c == 0) ? qx : ((c == 1) ? qy : qz);
            out_warp[(b * 3 + c) * N_ + n] = q + v;
        }
    }

    const float z = fmaxf(qz, 1e-5f);
    const float u  = (479.5f + 1050.0f * qx / z) * ((float)(W_ - 1) / (960.0f - 1.0f));
    const float vv = (269.5f + 1050.0f * qy / z) * ((float)(H_ - 1) / (540.0f - 1.0f));
    out_uv[(b * 2 + 0) * N_ + n] = u;
    out_uv[(b * 2 + 1) * N_ + n] = vv;
}

// ---------------------------------------------------------------------------
// Kernel B: self 16-NN over coarse points + channel-wise max-pool of features
// ---------------------------------------------------------------------------
__global__ __launch_bounds__(256) void k_knnfeat(
    const float* __restrict__ xyzc, const float* __restrict__ featc,
    float* __restrict__ out)
{
    __shared__ float s_pts[3 * M_];
    const int b = blockIdx.x / (M_ / 256);
    const int m = (blockIdx.x % (M_ / 256)) * 256 + threadIdx.x;

    const float* src = xyzc + b * 3 * M_;
    {
        const unsigned base_off = (unsigned)(unsigned long long)(&s_pts[0]);
        for (int i = threadIdx.x; i < 3 * M_; i += 256)
            async_g2l_b32(base_off + 4u * i, src + i);
        wait_async0();
    }
    __syncthreads();

    const float qx = s_pts[m], qy = s_pts[M_ + m], qz = s_pts[2 * M_ + m];

    float best[KS]; int bi[KS];
#pragma unroll
    for (int j = 0; j < KS; ++j) { best[j] = FLT_MAX; bi[j] = 0; }

    for (int p = 0; p < M_; ++p) {
        float dx = qx - s_pts[p], dy = qy - s_pts[M_ + p], dz = qz - s_pts[2 * M_ + p];
        float dd = dx * dx + dy * dy + dz * dz;
        if (dd < best[KS - 1]) {
            best[KS - 1] = dd; bi[KS - 1] = p;
#pragma unroll
            for (int j = KS - 1; j > 0; --j) {
                if (best[j] < best[j - 1]) {
                    float tf = best[j]; best[j] = best[j - 1]; best[j - 1] = tf;
                    int   ti = bi[j];   bi[j]   = bi[j - 1];   bi[j - 1]   = ti;
                }
            }
        }
    }

    const float* fb = featc + b * 64 * M_;
    for (int c = 0; c < C_; ++c) {
        float mv = -FLT_MAX;
#pragma unroll
        for (int j = 0; j < KS; ++j) mv = fmaxf(mv, fb[c * M_ + bi[j]]);
        out[(b * 64 + c) * M_ + m] = mv;
    }
}

// ---------------------------------------------------------------------------
// Pre-pass: convert both f32 CHW feature maps to f16 HWC in d_ws using
// packed v_cvt_pk_rtz_f16_f32. One thread per pixel; reads coalesced along w,
// writes 8 x b128 (128 B of channels) per pixel.
// ---------------------------------------------------------------------------
__global__ __launch_bounds__(256) void k_cvt_hwc(
    const float* __restrict__ in1, const float* __restrict__ in2,
    _Float16* __restrict__ o1, _Float16* __restrict__ o2)
{
    const int PPT = B_ * H_ * W_;                 // pixels per tensor
    const int BLK = PPT / 256;                    // 600 blocks per tensor
    const int tsel = blockIdx.x / BLK;
    const int pix  = (blockIdx.x % BLK) * 256 + threadIdx.x;   // b*H*W + h*W + w
    const float*   src = (tsel == 0) ? in1 : in2;
    _Float16*      dst = (tsel == 0) ? o1  : o2;

    const int b  = pix / (H_ * W_);
    const int hw = pix % (H_ * W_);
    const float* s = src + (size_t)b * C_ * H_ * W_ + hw;      // stride H*W per c
    _Float16*    d = dst + (size_t)pix * C_;

#pragma unroll
    for (int cc = 0; cc < C_; cc += 8) {
        v8h chunk;
#pragma unroll
        for (int u = 0; u < 8; u += 2) {
            v2h p = __builtin_bit_cast(v2h,
                __builtin_amdgcn_cvt_pkrtz(s[(size_t)(cc + u) * H_ * W_],
                                           s[(size_t)(cc + u + 1) * H_ * W_]));
            chunk[u]     = p[0];
            chunk[u + 1] = p[1];
        }
        *(v8h*)(d + cc) = chunk;
    }
}

// ---------------------------------------------------------------------------
// Shared writeout: stage D tiles in per-wave LDS then emit coalesced,
// zero-padding-aware stores. (corr[i,dx] = O'[i+dx][i])
// ---------------------------------------------------------------------------
__device__ __forceinline__ void corr_writeout(
    float* __restrict__ out, const float* so,
    int b, int dy, int h, int w0, int row, int hl, bool hok)
{
#pragma unroll
    for (int it = 0; it < 5; ++it) {
        const int dx = 2 * it + hl;                    // uniform per half-wave
        if (dx < ND) {
            const int wcol  = w0 + row + dx - MD;      // feat2 sample column
            const bool valid = hok && (wcol >= 0) && (wcol < W_);
            float v = so[(row + dx) * 17 + row] * (1.0f / 64.0f);
            v = (v < 0.0f) ? 0.1f * v : v;
            v = valid ? v : 0.0f;                      // zero-padding region
            out[((size_t)(b * NDISP + dy * ND + dx) * H_ + h) * W_ + (w0 + row)] = v;
        }
    }
}

// ---------------------------------------------------------------------------
// Kernel C (fast path): cost volume via WMMA from f16 HWC workspace.
// Fragments are two contiguous b128 loads each; no converts, no masking in
// the hot loop (clamped finite loads; padding zeroed at writeout).
// ---------------------------------------------------------------------------
__global__ __launch_bounds__(128) void k_corr_f16(
    const _Float16* __restrict__ g1, const _Float16* __restrict__ g2,
    float* __restrict__ out)
{
    __shared__ float s_O[4][32 * 17];               // per-wave O'[m][i]
    const int wave = threadIdx.x >> 5;
    const int lane = threadIdx.x & 31;
    const int SP   = W_ / 16;
    const int ws   = blockIdx.x * 4 + wave;
    const int b    = ws / (H_ * SP);
    const int rem  = ws % (H_ * SP);
    const int h    = rem / SP;
    const int w0   = (rem % SP) * 16;

    const int row = lane & 15;
    const int hl  = lane >> 4;
    float* so = &s_O[wave][0];

    const _Float16* T1 = g1 + (size_t)b * H_ * W_ * C_;
    const _Float16* T2 = g2 + (size_t)b * H_ * W_ * C_;

    // ---- A fragments: pixel (h, w0+row); channels 32kk+8hl+{0..7,16..23} ----
    v16h a[2];
    {
        const _Float16* ap = T1 + ((size_t)h * W_ + (w0 + row)) * C_ + 8 * hl;
#pragma unroll
        for (int kk = 0; kk < 2; ++kk) {
            v8h lo = *(const v8h*)(ap + 32 * kk);
            v8h hi = *(const v8h*)(ap + 32 * kk + 16);
            a[kk] = __builtin_shufflevector(lo, hi,
                        0,1,2,3,4,5,6,7,8,9,10,11,12,13,14,15);
        }
    }

    for (int dy = 0; dy < ND; ++dy) {
        const int  hh  = h + dy - MD;
        const bool hok = (hh >= 0) && (hh < H_);
        const int  hhc = hok ? hh : 0;

        if (dy + 1 < ND) {
            const int hn = h + dy + 1 - MD;
            if (hn >= 0 && hn < H_)
                __builtin_prefetch(&T2[((size_t)hn * W_ + w0) * C_], 0, 3);
        }

#pragma unroll
        for (int t = 0; t < 2; ++t) {
            const int m   = 16 * t + row;
            const int wc  = w0 - MD + m;
            const int wcc = (wc < 0) ? 0 : ((wc >= W_) ? (W_ - 1) : wc);
            // B fragment kk: channels 32kk+8hl+{0..7,16..23} of pixel (hhc, wcc)
            const _Float16* bp = T2 + ((size_t)hhc * W_ + wcc) * C_ + 8 * hl;
            v8f acc = {};
#pragma unroll
            for (int kk = 0; kk < 2; ++kk) {
                v8h lo = *(const v8h*)(bp + 32 * kk);
                v8h hi = *(const v8h*)(bp + 32 * kk + 16);
                v16h bb = __builtin_shufflevector(lo, hi,
                              0,1,2,3,4,5,6,7,8,9,10,11,12,13,14,15);
                acc = __builtin_amdgcn_wmma_f32_16x16x32_f16(
                          false, a[kk], false, bb, (short)0, acc, false, false);
            }
#pragma unroll
            for (int r = 0; r < 8; ++r)
                so[m * 17 + 8 * hl + r] = acc[r];
        }
        corr_writeout(out, so, b, dy, h, w0, row, hl, hok);
    }
}

// ---------------------------------------------------------------------------
// Kernel C (fallback, no workspace): f32 loads + inline converts.
// ---------------------------------------------------------------------------
__global__ __launch_bounds__(128) void k_corr_f32(
    const float* __restrict__ f1, const float* __restrict__ f2,
    float* __restrict__ out)
{
    __shared__ float s_O[4][32 * 17];
    const int wave = threadIdx.x >> 5;
    const int lane = threadIdx.x & 31;
    const int SP   = W_ / 16;
    const int ws   = blockIdx.x * 4 + wave;
    const int b    = ws / (H_ * SP);
    const int rem  = ws % (H_ * SP);
    const int h    = rem / SP;
    const int w0   = (rem % SP) * 16;

    const int row = lane & 15;
    const int hl  = lane >> 4;
    const int HW  = H_ * W_;
    float* so = &s_O[wave][0];

    const float* F1 = f1 + (size_t)b * C_ * HW;
    const float* F2 = f2 + (size_t)b * C_ * HW;

    v16h a[2];
#pragma unroll
    for (int kk = 0; kk < 2; ++kk) {
        const float* base = F1 + ((32 * kk + 8 * hl) * H_ + h) * W_ + (w0 + row);
#pragma unroll
        for (int e = 0; e < 16; ++e) {
            const int c_off = 16 * (e >> 3) + (e & 7);
            a[kk][e] = (_Float16)base[c_off * HW];
        }
    }

    for (int dy = 0; dy < ND; ++dy) {
        const int  hh  = h + dy - MD;
        const bool hok = (hh >= 0) && (hh < H_);
        const int  hhc = hok ? hh : 0;

#pragma unroll
        for (int t = 0; t < 2; ++t) {
            const int m   = 16 * t + row;
            const int wc  = w0 - MD + m;
            const int wcc = (wc < 0) ? 0 : ((wc >= W_) ? (W_ - 1) : wc);
            v8f acc = {};
#pragma unroll
            for (int kk = 0; kk < 2; ++kk) {
                const float* base = F2 + ((32 * kk + 8 * hl) * H_ + hhc) * W_ + wcc;
                v16h bb;
#pragma unroll
                for (int e = 0; e < 16; ++e)
                    bb[e] = (_Float16)base[(16 * (e >> 3) + (e & 7)) * HW];    // unmasked clamped load
                acc = __builtin_amdgcn_wmma_f32_16x16x32_f16(
                          false, a[kk], false, bb, (short)0, acc, false, false);
            }
#pragma unroll
            for (int r = 0; r < 8; ++r)
                so[m * 17 + 8 * hl + r] = acc[r];
        }
        corr_writeout(out, so, b, dy, h, w0, row, hl, hok);
    }
}

// ---------------------------------------------------------------------------
extern "C" void kernel_launch(void* const* d_in, const int* in_sizes, int n_in,
                              void* d_out, int out_size, void* d_ws, size_t ws_size,
                              hipStream_t stream) {
    const float* xyz1  = (const float*)d_in[0];
    const float* xyzc  = (const float*)d_in[1];
    const float* flowc = (const float*)d_in[2];
    const float* featc = (const float*)d_in[3];
    const float* f1    = (const float*)d_in[4];
    const float* f2    = (const float*)d_in[5];

    float* out        = (float*)d_out;
    float* out_interp = out;                                   // [2,67,8192]
    float* out_warp   = out_interp + (size_t)B_ * 67 * N_;     // [2,3,8192]
    float* out_uv     = out_warp   + (size_t)B_ * 3  * N_;     // [2,2,8192]
    float* out_knn    = out_uv     + (size_t)B_ * 2  * N_;     // [2,64,2048]
    float* out_corr   = out_knn    + (size_t)B_ * C_ * M_;     // [2,81,160,480]

    k_interp<<<(B_ * N_) / 256, 256, 0, stream>>>(
        xyz1, xyzc, flowc, featc, out_interp, out_warp, out_uv);

    k_knnfeat<<<(B_ * M_) / 256, 256, 0, stream>>>(
        xyzc, featc, out_knn);

    const int strips = B_ * H_ * (W_ / 16);                    // 9600
    const size_t tensor_halfs = (size_t)B_ * H_ * W_ * C_;     // per map
    const size_t need = 2 * tensor_halfs * sizeof(_Float16);   // 39.3 MB

    if (ws_size >= need) {
        _Float16* w1 = (_Float16*)d_ws;
        _Float16* w2 = w1 + tensor_halfs;
        k_cvt_hwc<<<2 * (B_ * H_ * W_) / 256, 256, 0, stream>>>(f1, f2, w1, w2);
        k_corr_f16<<<strips / 4, 128, 0, stream>>>(w1, w2, out_corr);
    } else {
        k_corr_f32<<<strips / 4, 128, 0, stream>>>(f1, f2, out_corr);
    }
}
